// CalculateSLayer_58471684767748
// MI455X (gfx1250) — matrix operand order repaired and checked by reading, not compiled, not run on hardware
//
#include <hip/hip_runtime.h>
#include <hip/hip_bf16.h>

// ---------------------------------------------------------------------------
// h_in  = A^T h,  h_out = A h,  A = adj.sum(edge axis),  L=4096, D=150.
// Two bf16-WMMA GEMM passes (one per output), each streaming adj (134MB) once.
// HBM-bound: ~268MB @ 23.3 TB/s ~= 11.5us floor; WMMA compute ~1-2us.
// ---------------------------------------------------------------------------

typedef __attribute__((ext_vector_type(16))) __bf16 v16bf;
typedef __attribute__((ext_vector_type(8)))  __bf16 v8bf;
typedef __attribute__((ext_vector_type(8)))  float  v8f;
typedef __attribute__((ext_vector_type(2)))  float  f32x2;

#define LENG   4096
#define DOUT   150
#define NT     10            // 10 n-tiles of 16 -> N padded to 160
#define BM     64            // rows per workgroup (4 waves x 16)
#define KSTEPS (LENG / 32)   // 128 k-steps of 32

// ---------------------------------------------------------------------------
// Kernel 1: h (f32 [4096][150]) -> bf16 B-fragments, pre-swizzled to the
// v_wmma_f32_16x16x32_bf16 B operand in contiguous k order (same as A):
//   fragment (kb, nt): 32 lanes x 16 bf16 (32B) contiguous per lane;
//   lane l: n = nt*16 + (l&15), element e: k = kb*32 + (l>>4)*16 + e.
// ---------------------------------------------------------------------------
__global__ void pack_h_kernel(const float* __restrict__ h,
                              __bf16* __restrict__ hf) {
  int t = blockIdx.x * blockDim.x + threadIdx.x;     // t == (kb*NT + nt)*32 + lane
  if (t >= KSTEPS * NT * 32) return;
  int lane = t & 31;
  int nt   = (t >> 5) % NT;
  int kb   = t / (32 * NT);
  int n    = nt * 16 + (lane & 15);
  int g    = lane >> 4;
  v16bf v;
#pragma unroll
  for (int e = 0; e < 16; ++e) {
    int   k = kb * 32 + g * 16 + e;
    float x = (n < DOUT) ? h[(size_t)k * DOUT + n] : 0.0f;
    v[e] = (__bf16)x;
  }
  *(v16bf*)(hf + (size_t)t * 16) = v;
}

// ---------------------------------------------------------------------------
// Kernel 2 (templated on TRANS): OUT[m,d] = sum_k Abf[m,k] * hb[k,d]
//   TRANS=1: Abf[m,k] = adj[k,m,0]+adj[k,m,1]  (h_in)
//   TRANS=0: Abf[m,k] = adj[m,k,0]+adj[m,k,1]  (h_out)
// 128 threads = 4 waves; each wave owns 16 rows; N=160 in 10 f32 accum tiles.
// B fragments preloaded into registers before the WMMA burst so the 20 b128
// loads issue as one clause instead of load->wait->wmma chains.
// ---------------------------------------------------------------------------
template <int TRANS>
__global__ __launch_bounds__(128)
void gemm_bf16_kernel(const float* __restrict__ adj,
                      const __bf16* __restrict__ hf,
                      float* __restrict__ out) {
  // row padded 32 -> 40 elems (80B) to spread LDS banks on fragment reads
  __shared__ alignas(16) __bf16 As[BM][40];

  const int tid  = threadIdx.x;
  const int lane = tid & 31;
  const int wave = tid >> 5;
  const int m0   = blockIdx.x * BM;

  v8f acc[NT];
#pragma unroll
  for (int i = 0; i < NT; ++i)
#pragma unroll
    for (int r = 0; r < 8; ++r) acc[i][r] = 0.0f;

  const int g  = lane >> 4;   // k-half selector
  const int mr = lane & 15;   // row within 16-row wmma tile

  for (int kb = 0; kb < KSTEPS; ++kb) {
    const int k0 = kb * 32;

    // ---- prefetch next adj tile (global_prefetch_b8) ----
    if (kb + 1 < KSTEPS) {
      const int kn = k0 + 32;
      if (!TRANS) {
        int row = tid >> 1, half = tid & 1;          // 64 rows x 256B
        __builtin_prefetch(adj + (((size_t)(m0 + row)) * LENG + kn) * 2 + half * 32, 0, 0);
      } else {
        int row = tid >> 2, part = tid & 3;          // 32 rows x 512B
        __builtin_prefetch(adj + (((size_t)(kn + row)) * LENG + m0) * 2 + part * 32, 0, 0);
      }
    }

    __syncthreads();
    // ---- cooperative stage of 64x32 A tile (sum 2 edge chans, cvt bf16) ----
    if (!TRANS) {
      // adj element (m,k) at ((m0+m)*L + k0+k)*2 ; threads sweep k fast
      const int kk = tid & 31, mb = tid >> 5;        // 4 rows/pass, 16 passes
#pragma unroll
      for (int s = 0; s < 16; ++s) {
        int   m  = mb + s * 4;
        f32x2 pv = *(const f32x2*)(adj + (((size_t)(m0 + m)) * LENG + (k0 + kk)) * 2);
        As[m][kk] = (__bf16)(pv[0] + pv[1]);
      }
    } else {
      // adj element (m,k) at ((k0+k)*L + m0+m)*2 ; threads sweep m fast
      const int mm = tid & 63, kbse = tid >> 6;      // 2 k-rows/pass, 16 passes
#pragma unroll
      for (int s = 0; s < 16; ++s) {
        int   k  = kbse + s * 2;
        f32x2 pv = *(const f32x2*)(adj + (((size_t)(k0 + k)) * LENG + (m0 + mm)) * 2);
        As[mm][k] = (__bf16)(pv[0] + pv[1]);
      }
    }
    __syncthreads();

    // ---- A fragment: 16x32, contiguous k order (k = g*16 + e), two ds_load_b128 ----
    union { v16bf v; v8bf h[2]; } af;
    af.h[0] = *(const v8bf*)(&As[wave * 16 + mr][g * 16]);       // K = g*16 .. g*16+7
    af.h[1] = *(const v8bf*)(&As[wave * 16 + mr][g * 16 + 8]);   // K = g*16+8 ..

    // ---- preload all 10 B fragments (one clause of b128s from L2) ----
    const __bf16* bb = hf + ((size_t)kb * NT * 32 + lane) * 16;
    v16bf bfr[NT];
#pragma unroll
    for (int nt = 0; nt < NT; ++nt)
      bfr[nt] = *(const v16bf*)(bb + (size_t)nt * 32 * 16);

    // ---- WMMA burst ----
#pragma unroll
    for (int nt = 0; nt < NT; ++nt)
      acc[nt] = __builtin_amdgcn_wmma_f32_16x16x32_bf16(
          false, af.v, false, bfr[nt], (short)0, acc[nt], false, false);
  }

  // ---- store: C/D layout VGPR r -> row g*8+r, col = lane&15 ----
  const int nn = lane & 15;
#pragma unroll
  for (int nt = 0; nt < NT; ++nt) {
    int col = nt * 16 + nn;
    if (col >= DOUT) continue;
#pragma unroll
    for (int r = 0; r < 8; ++r) {
      int row = m0 + wave * 16 + g * 8 + r;
      out[(size_t)row * DOUT + col] = acc[nt][r];
    }
  }
}

extern "C" void kernel_launch(void* const* d_in, const int* in_sizes, int n_in,
                              void* d_out, int out_size, void* d_ws, size_t ws_size,
                              hipStream_t stream) {
  const float* adj = (const float*)d_in[0];   // [4096,4096,2] f32
  const float* h   = (const float*)d_in[1];   // [4096,150]    f32
  float* out       = (float*)d_out;           // h_in ++ h_out, each [4096,150] f32
  __bf16* hf       = (__bf16*)d_ws;           // 128*10*32*16 bf16 = 1.31 MB

  const int packN = KSTEPS * NT * 32;         // 40960 threads
  pack_h_kernel<<<dim3((packN + 255) / 256), dim3(256), 0, stream>>>(h, hf);

  dim3 grid(LENG / BM, 1, 1);
  // h_in = A^T h  -> first half of d_out
  gemm_bf16_kernel<1><<<grid, dim3(128), 0, stream>>>(adj, hf, out);
  // h_out = A h   -> second half of d_out
  gemm_bf16_kernel<0><<<grid, dim3(128), 0, stream>>>(adj, hf, out + (size_t)LENG * DOUT);
}
